// TransformerBlock_52793738002968
// MI455X (gfx1250) — compile-verified, matrix operand order checked by reading, not run on hardware
//
#include <hip/hip_runtime.h>
#include <hip/hip_bf16.h>

// ---------------------------------------------------------------------------
// Transformer block forward for MI455X (gfx1250, wave32, WMMA).
// GEMMs: v_wmma_f32_16x16x32_bf16 + async global->LDS double buffering.
// ---------------------------------------------------------------------------

typedef __bf16 bf16_t;
typedef __attribute__((ext_vector_type(16))) __bf16 v16bf;
typedef __attribute__((ext_vector_type(8)))  __bf16 v8bf;
typedef __attribute__((ext_vector_type(8)))  float  v8f;

#define WMMA_BF16(a, b, c) \
  __builtin_amdgcn_wmma_f32_16x16x32_bf16(false, (a), false, (b), (short)0, (c), false, false)

static constexpr int kB  = 2;
static constexpr int kT  = 2048;
static constexpr int kC  = 1024;
static constexpr int kNH = 16;
static constexpr int kHD = 64;       // kC / kNH
static constexpr int kFF = 4096;     // 4 * kC
static constexpr int kTok = kB * kT; // 4096 rows

// ---------------------------------------------------------------------------
// Async global -> LDS copy (CDNA5 TDM-adjacent path, ISA 15.18.3 op 98).
// VDST = VGPR holding LDS byte address (low 32 bits of flat LDS pointer);
// completion tracked by ASYNCcnt.
// ---------------------------------------------------------------------------
__device__ inline void async_copy_b128(void* lds, const void* gptr) {
  unsigned l = (unsigned)(unsigned long long)lds;   // flat[31:0] == LDS offset
  unsigned long long g = (unsigned long long)gptr;
  asm volatile("global_load_async_to_lds_b128 %0, %1, off"
               :: "v"(l), "v"(g) : "memory");
}
__device__ inline void wait_async0() {
  asm volatile("s_wait_asynccnt 0x0" ::: "memory");
}

// ---------------------------------------------------------------------------
// Fragment loader. Tile in LDS is row-major [rows][stride] bf16.
// A-matrix 16x32 layout (ISA 7.12.2): lanes 0-15 -> K {0..7, 16..23},
// lanes 16-31 -> K {8..15, 24..31}; two contiguous 128-bit LDS loads.
// Same pattern serves B fragments when the tile is stored [N][K].
// ---------------------------------------------------------------------------
__device__ inline v16bf ld_frag(const bf16_t* s, int stride, int row0, int k0) {
  int lane = threadIdx.x & 31;
  int r    = lane & 15;
  int hi   = lane >> 4;
  const bf16_t* p = s + (row0 + r) * stride + k0 + hi * 8;
  v8bf lo  = *(const v8bf*)(p);
  v8bf hiv = *(const v8bf*)(p + 16);
  return __builtin_shufflevector(lo, hiv, 0,1,2,3,4,5,6,7,8,9,10,11,12,13,14,15);
}

// Pack 4 floats -> 4 bf16 -> one 64-bit store (contiguous along K).
__device__ inline void st_pack4(bf16_t* dst, float a, float b, float c, float d) {
  union { bf16_t h[4]; uint2 u; } p;
  p.h[0] = (bf16_t)a; p.h[1] = (bf16_t)b; p.h[2] = (bf16_t)c; p.h[3] = (bf16_t)d;
  *(uint2*)dst = p.u;
}
__device__ inline void st_pack4bf(bf16_t* dst, bf16_t a, bf16_t b, bf16_t c, bf16_t d) {
  union { bf16_t h[4]; uint2 u; } p;
  p.h[0] = a; p.h[1] = b; p.h[2] = c; p.h[3] = d;
  *(uint2*)dst = p.u;
}

// ---------------------------------------------------------------------------
// Weight prep: in[K][N] fp32 -> out[N][K] bf16 (transpose + convert), once
// per launch so the GEMMs stream bf16 rows with plain/async B128 copies.
// ---------------------------------------------------------------------------
__global__ __launch_bounds__(256) void wprep_kernel(const float* __restrict__ in,
                                                    bf16_t* __restrict__ out,
                                                    int K, int N) {
  __shared__ float t[64][65];
  const int k0 = blockIdx.y * 64, n0 = blockIdx.x * 64;
  const int tid = threadIdx.x;
#pragma unroll
  for (int c = 0; c < 4; ++c) {
    int idx = tid + c * 256;
    int r = idx >> 4, n = (idx & 15) * 4;
    float4 f = *(const float4*)&in[(size_t)(k0 + r) * N + n0 + n];
    t[r][n + 0] = f.x; t[r][n + 1] = f.y; t[r][n + 2] = f.z; t[r][n + 3] = f.w;
  }
  __syncthreads();
  const int r = tid >> 2, seg = (tid & 3) * 16;
  bf16_t* orow = out + (size_t)(n0 + r) * K + k0 + seg;
#pragma unroll
  for (int q = 0; q < 4; ++q)
    st_pack4(&orow[q * 4], t[seg + q * 4 + 0][r], t[seg + q * 4 + 1][r],
                           t[seg + q * 4 + 2][r], t[seg + q * 4 + 3][r]);
}

// ---------------------------------------------------------------------------
// LayerNorm: fp32 row [C] -> normalized bf16 row. One block per row.
// ---------------------------------------------------------------------------
__global__ __launch_bounds__(256) void ln_kernel(const float* __restrict__ x,
                                                 const float* __restrict__ w,
                                                 const float* __restrict__ b,
                                                 bf16_t* __restrict__ out) {
  const int row = blockIdx.x;
  const int tid = threadIdx.x;
  const float* xr = x + (size_t)row * kC;
  float4 f = *(const float4*)&xr[tid * 4];

  __shared__ float red[8];
  float s = f.x + f.y + f.z + f.w;
  for (int m = 16; m >= 1; m >>= 1) s += __shfl_xor(s, m, 32);
  if ((tid & 31) == 0) red[tid >> 5] = s;
  __syncthreads();
  float mean = 0.f;
  for (int i = 0; i < 8; ++i) mean += red[i];
  mean *= (1.0f / kC);
  __syncthreads();

  float d0 = f.x - mean, d1 = f.y - mean, d2 = f.z - mean, d3 = f.w - mean;
  float vs = d0 * d0 + d1 * d1 + d2 * d2 + d3 * d3;
  for (int m = 16; m >= 1; m >>= 1) vs += __shfl_xor(vs, m, 32);
  if ((tid & 31) == 0) red[tid >> 5] = vs;
  __syncthreads();
  float var = 0.f;
  for (int i = 0; i < 8; ++i) var += red[i];
  var *= (1.0f / kC);
  const float inv = rsqrtf(var + 1e-5f);

  const int c = tid * 4;
  bf16_t* o = out + (size_t)row * kC + c;
  union { bf16_t h[4]; uint2 u; } p;
  p.h[0] = (bf16_t)(d0 * inv * w[c + 0] + b[c + 0]);
  p.h[1] = (bf16_t)(d1 * inv * w[c + 1] + b[c + 1]);
  p.h[2] = (bf16_t)(d2 * inv * w[c + 2] + b[c + 2]);
  p.h[3] = (bf16_t)(d3 * inv * w[c + 3] + b[c + 3]);
  *(uint2*)o = p.u;
}

// ---------------------------------------------------------------------------
// GEMM: Y = act( X[M,K](bf16) @ Wt[N,K](bf16, pre-transposed) + bias )(+res)
// Block tile 128x128, BK=64, 256 threads = 8 waves (2x4), wave tile 64x32.
// Double-buffered async global->LDS copies overlap HBM with WMMA.
// ---------------------------------------------------------------------------
template <int N, int K, bool BIAS, bool RES, bool RELU, bool OUTF, bool OUTB>
__global__ __launch_bounds__(256) void gemm_bf16_kernel(
    const bf16_t* __restrict__ X, const bf16_t* __restrict__ Wt,
    const float* __restrict__ bias, const float* __restrict__ resid,
    float* __restrict__ outF, bf16_t* __restrict__ outB) {
  __shared__ bf16_t sA[2][128 * 64];   // [m][k]
  __shared__ bf16_t sB[2][128 * 64];   // [n][k]

  const int tid  = threadIdx.x;
  const int lane = tid & 31;
  const int wave = tid >> 5;
  const int wm   = (wave >> 2) * 64;  // 0 or 64
  const int wn   = (wave & 3) * 32;   // 0,32,64,96
  const int m0   = blockIdx.y * 128;
  const int n0   = blockIdx.x * 128;

  v8f acc[4][2];
#pragma unroll
  for (int i = 0; i < 4; ++i)
#pragma unroll
    for (int j = 0; j < 2; ++j)
      acc[i][j] = (v8f){0.f,0.f,0.f,0.f,0.f,0.f,0.f,0.f};

  // Per-thread copy slots: 4 x B128 for A, 4 x B128 for B per tile.
  auto issue_tile = [&](int buf, int k0) {
#pragma unroll
    for (int c = 0; c < 4; ++c) {
      int idx = tid + c * 256;
      int r = idx >> 3, kc = (idx & 7) * 8;
      async_copy_b128(&sA[buf][r * 64 + kc], &X[(size_t)(m0 + r) * K + k0 + kc]);
      async_copy_b128(&sB[buf][r * 64 + kc], &Wt[(size_t)(n0 + r) * K + k0 + kc]);
    }
  };

  constexpr int NK = K / 64;
  issue_tile(0, 0);
  for (int i = 0; i < NK; ++i) {
    const int cur = i & 1;
    wait_async0();        // this wave's tile i landed in LDS
    __syncthreads();      // all waves landed tile i; all done reading buf[1-cur]
    if (i + 1 < NK) issue_tile(1 - cur, (i + 1) * 64);

    const bf16_t* A = sA[cur];
    const bf16_t* Bt = sB[cur];
#pragma unroll
    for (int kk = 0; kk < 64; kk += 32) {
      v16bf afr[4], bfr[2];
#pragma unroll
      for (int ii = 0; ii < 4; ++ii) afr[ii] = ld_frag(A, 64, wm + ii * 16, kk);
#pragma unroll
      for (int jj = 0; jj < 2; ++jj) bfr[jj] = ld_frag(Bt, 64, wn + jj * 16, kk);
#pragma unroll
      for (int ii = 0; ii < 4; ++ii)
#pragma unroll
        for (int jj = 0; jj < 2; ++jj)
          acc[ii][jj] = WMMA_BF16(afr[ii], bfr[jj], acc[ii][jj]);
    }
  }

  // Epilogue. C/D layout: lane col = lane&15, row = v + 8*(lane>>4).
  const int cn = lane & 15, hi = lane >> 4;
#pragma unroll
  for (int i = 0; i < 4; ++i) {
    const int row0 = m0 + wm + i * 16 + 8 * hi;
#pragma unroll
    for (int j = 0; j < 2; ++j) {
      const int col = n0 + wn + j * 16 + cn;
      float bv = 0.f;
      if constexpr (BIAS) bv = bias[col];
#pragma unroll
      for (int v = 0; v < 8; ++v) {
        const size_t off = (size_t)(row0 + v) * N + col;
        float val = acc[i][j][v] + bv;
        if constexpr (RELU) val = fmaxf(val, 0.f);
        if constexpr (RES) val += resid[off];
        if constexpr (OUTF) outF[off] = val;
        if constexpr (OUTB) outB[off] = (bf16_t)val;
      }
    }
  }
}

// ---------------------------------------------------------------------------
// Flash-style causal attention. One block (4 waves, 128 thr) per
// (64-query tile, batch*head). Each wave owns 16 query rows.
// Q/K/V are bf16 [B,T,C] with per-head slice at h*HD.
// ---------------------------------------------------------------------------
__global__ __launch_bounds__(128) void attn_kernel(const bf16_t* __restrict__ Q,
                                                   const bf16_t* __restrict__ Kg,
                                                   const bf16_t* __restrict__ V,
                                                   bf16_t* __restrict__ O) {
  __shared__ bf16_t sQ[64 * 64];       // [q][d]
  __shared__ bf16_t sK[64 * 64];       // [key][d]      -> B frag for S=Q*K^T
  __shared__ bf16_t sVt[64 * 64];      // [d][key]      -> B frag for O=P*V
  __shared__ bf16_t sP[4][16 * 64];    // per-wave P tile for layout transpose

  const int tid  = threadIdx.x;
  const int lane = tid & 31;
  const int wave = tid >> 5;
  const int bh   = blockIdx.y;
  const int bidx = bh / kNH, h = bh % kNH;
  const int qt   = blockIdx.x;
  const int t0   = qt * 64;
  const size_t base = (size_t)bidx * kT * kC + (size_t)h * kHD;
  const float scale = 0.125f;  // 1/sqrt(64)

  // Q tile (64x64 bf16) via async copies.
#pragma unroll
  for (int c = 0; c < 4; ++c) {
    int idx = tid + c * 128;
    int r = idx >> 3, dc = (idx & 7) * 8;
    async_copy_b128(&sQ[r * 64 + dc], &Q[base + (size_t)(t0 + r) * kC + dc]);
  }
  wait_async0();
  __syncthreads();
  v16bf qfr0 = ld_frag(sQ, 64, wave * 16, 0);
  v16bf qfr1 = ld_frag(sQ, 64, wave * 16, 32);

  float mrow[8], lrow[8];
  v8f o[4];
#pragma unroll
  for (int v = 0; v < 8; ++v) { mrow[v] = -3.0e38f; lrow[v] = 0.f; }
#pragma unroll
  for (int dt = 0; dt < 4; ++dt)
    o[dt] = (v8f){0.f,0.f,0.f,0.f,0.f,0.f,0.f,0.f};

  for (int j = 0; j <= qt; ++j) {
    __syncthreads();
    // K tile natural [key][d] via async copies.
#pragma unroll
    for (int c = 0; c < 4; ++c) {
      int idx = tid + c * 128;
      int r = idx >> 3, dc = (idx & 7) * 8;
      async_copy_b128(&sK[r * 64 + dc], &Kg[base + (size_t)(j * 64 + r) * kC + dc]);
    }
    // V transposed into [d][key]: 4-key x 8-d block -> 8 packed ds_store_b64.
    {
      const int r  = (tid >> 3) * 4;
      const int dc = (tid & 7) * 8;
      uint4 u0 = *(const uint4*)&V[base + (size_t)(j * 64 + r + 0) * kC + dc];
      uint4 u1 = *(const uint4*)&V[base + (size_t)(j * 64 + r + 1) * kC + dc];
      uint4 u2 = *(const uint4*)&V[base + (size_t)(j * 64 + r + 2) * kC + dc];
      uint4 u3 = *(const uint4*)&V[base + (size_t)(j * 64 + r + 3) * kC + dc];
      const bf16_t* e0 = (const bf16_t*)&u0;
      const bf16_t* e1 = (const bf16_t*)&u1;
      const bf16_t* e2 = (const bf16_t*)&u2;
      const bf16_t* e3 = (const bf16_t*)&u3;
#pragma unroll
      for (int t = 0; t < 8; ++t)
        st_pack4bf(&sVt[(dc + t) * 64 + r], e0[t], e1[t], e2[t], e3[t]);
    }
    wait_async0();
    __syncthreads();

    // S = Q K^T (16x64 per wave, chained K=32 WMMAs).
    v8f s[4];
#pragma unroll
    for (int nt = 0; nt < 4; ++nt) {
      v8f z = (v8f){0.f,0.f,0.f,0.f,0.f,0.f,0.f,0.f};
      v16bf kf0 = ld_frag(sK, 64, nt * 16, 0);
      v16bf kf1 = ld_frag(sK, 64, nt * 16, 32);
      z = WMMA_BF16(qfr0, kf0, z);
      z = WMMA_BF16(qfr1, kf1, z);
      s[nt] = z;
    }

    const int cn = lane & 15, hi = lane >> 4;
    float rmax[8];
#pragma unroll
    for (int v = 0; v < 8; ++v) rmax[v] = -3.0e38f;
#pragma unroll
    for (int nt = 0; nt < 4; ++nt) {
      const int ng = j * 64 + nt * 16 + cn;
#pragma unroll
      for (int v = 0; v < 8; ++v) {
        const int mg = t0 + wave * 16 + v + 8 * hi;
        float val = s[nt][v] * scale;
        if (ng > mg) val = -3.0e38f;           // causal mask
        s[nt][v] = val;
        rmax[v] = fmaxf(rmax[v], val);
      }
    }
    // Row reductions stay inside each 16-lane N-group (xor masks 1..8).
#pragma unroll
    for (int v = 0; v < 8; ++v)
      for (int m = 1; m < 16; m <<= 1)
        rmax[v] = fmaxf(rmax[v], __shfl_xor(rmax[v], m, 32));

    float alpha[8], rsum[8];
#pragma unroll
    for (int v = 0; v < 8; ++v) {
      const float mn = fmaxf(mrow[v], rmax[v]);
      alpha[v] = __expf(mrow[v] - mn);
      mrow[v]  = mn;
      rsum[v]  = 0.f;
    }

    bf16_t* sPw = &sP[wave][0];
#pragma unroll
    for (int nt = 0; nt < 4; ++nt) {
#pragma unroll
      for (int v = 0; v < 8; ++v) {
        const float p = __expf(s[nt][v] - mrow[v]);
        rsum[v] += p;
        sPw[(v + 8 * hi) * 64 + nt * 16 + cn] = (bf16_t)p;  // C-layout -> [m][k]
      }
    }
#pragma unroll
    for (int v = 0; v < 8; ++v) {
      float rs = rsum[v];
      for (int m = 1; m < 16; m <<= 1) rs += __shfl_xor(rs, m, 32);
      lrow[v] = lrow[v] * alpha[v] + rs;
    }
#pragma unroll
    for (int dt = 0; dt < 4; ++dt)
#pragma unroll
      for (int v = 0; v < 8; ++v) o[dt][v] *= alpha[v];
    __syncthreads();

    // O += P V
    v16bf pf0 = ld_frag(sPw, 64, 0, 0);
    v16bf pf1 = ld_frag(sPw, 64, 0, 32);
#pragma unroll
    for (int dt = 0; dt < 4; ++dt) {
      v16bf vf0 = ld_frag(sVt, 64, dt * 16, 0);
      v16bf vf1 = ld_frag(sVt, 64, dt * 16, 32);
      o[dt] = WMMA_BF16(pf0, vf0, o[dt]);
      o[dt] = WMMA_BF16(pf1, vf1, o[dt]);
    }
  }

  // Epilogue: O / l, write bf16 directly into [B,T,C] head slice.
  const int cn = lane & 15, hi = lane >> 4;
#pragma unroll
  for (int v = 0; v < 8; ++v) {
    const float inv = 1.0f / lrow[v];
    const int mg = t0 + wave * 16 + v + 8 * hi;
#pragma unroll
    for (int dt = 0; dt < 4; ++dt)
      O[base + (size_t)mg * kC + dt * 16 + cn] = (bf16_t)(o[dt][v] * inv);
  }
}

// ---------------------------------------------------------------------------
// Host-side orchestration.
// ---------------------------------------------------------------------------
extern "C" void kernel_launch(void* const* d_in, const int* in_sizes, int n_in,
                              void* d_out, int out_size, void* d_ws, size_t ws_size,
                              hipStream_t stream) {
  const float* x    = (const float*)d_in[0];
  const float* ln1w = (const float*)d_in[1];
  const float* ln1b = (const float*)d_in[2];
  const float* Wq   = (const float*)d_in[3];
  const float* Wk   = (const float*)d_in[4];
  const float* Wv   = (const float*)d_in[5];
  const float* Wp   = (const float*)d_in[6];
  const float* bp   = (const float*)d_in[7];
  const float* ln2w = (const float*)d_in[8];
  const float* ln2b = (const float*)d_in[9];
  const float* W1   = (const float*)d_in[10];
  const float* b1   = (const float*)d_in[11];
  const float* W2   = (const float*)d_in[12];
  const float* b2   = (const float*)d_in[13];
  float* out = (float*)d_out;

  const size_t MB = (size_t)1 << 20;
  char* ws = (char*)d_ws;
  bf16_t* h_bf  = (bf16_t*)(ws + 0 * MB);    // 4096x1024 bf16 =  8 MB
  bf16_t* q_bf  = (bf16_t*)(ws + 8 * MB);    //                   8 MB
  bf16_t* k_bf  = (bf16_t*)(ws + 16 * MB);   //                   8 MB
  bf16_t* v_bf  = (bf16_t*)(ws + 24 * MB);   //                   8 MB
  bf16_t* o_bf  = (bf16_t*)(ws + 32 * MB);   //                   8 MB
  float*  x1    = (float*) (ws + 40 * MB);   // 4096x1024 f32  = 16 MB
  bf16_t* h2_bf = (bf16_t*)(ws + 56 * MB);   //                   8 MB
  bf16_t* a1_bf = (bf16_t*)(ws + 64 * MB);   // 4096x4096 bf16 = 32 MB
  bf16_t* Wqt   = (bf16_t*)(ws + 96 * MB);   // [1024,1024] bf16  2 MB
  bf16_t* Wkt   = (bf16_t*)(ws + 98 * MB);   //                   2 MB
  bf16_t* Wvt   = (bf16_t*)(ws + 100 * MB);  //                   2 MB
  bf16_t* Wpt   = (bf16_t*)(ws + 102 * MB);  //                   2 MB
  bf16_t* W1t   = (bf16_t*)(ws + 104 * MB);  // [4096,1024]       8 MB
  bf16_t* W2t   = (bf16_t*)(ws + 112 * MB);  // [1024,4096]       8 MB -> 120 MB

  const dim3 blk256(256), blk128(128);
  const dim3 gC(kC / 128, kTok / 128);   // N=1024
  const dim3 gF(kFF / 128, kTok / 128);  // N=4096

  // 0) Weight prep: fp32 [K,N] -> bf16 [N,K], once per launch.
  wprep_kernel<<<dim3(kC / 64, kC / 64), blk256, 0, stream>>>(Wq, Wqt, kC, kC);
  wprep_kernel<<<dim3(kC / 64, kC / 64), blk256, 0, stream>>>(Wk, Wkt, kC, kC);
  wprep_kernel<<<dim3(kC / 64, kC / 64), blk256, 0, stream>>>(Wv, Wvt, kC, kC);
  wprep_kernel<<<dim3(kC / 64, kC / 64), blk256, 0, stream>>>(Wp, Wpt, kC, kC);
  wprep_kernel<<<dim3(kFF / 64, kC / 64), blk256, 0, stream>>>(W1, W1t, kC, kFF);
  wprep_kernel<<<dim3(kC / 64, kFF / 64), blk256, 0, stream>>>(W2, W2t, kFF, kC);

  // 1) h = LN1(x) -> bf16
  ln_kernel<<<kTok, blk256, 0, stream>>>(x, ln1w, ln1b, h_bf);
  // 2) q,k,v = h @ W{q,k,v} -> bf16
  gemm_bf16_kernel<kC, kC, false, false, false, false, true>
      <<<gC, blk256, 0, stream>>>(h_bf, Wqt, nullptr, nullptr, nullptr, q_bf);
  gemm_bf16_kernel<kC, kC, false, false, false, false, true>
      <<<gC, blk256, 0, stream>>>(h_bf, Wkt, nullptr, nullptr, nullptr, k_bf);
  gemm_bf16_kernel<kC, kC, false, false, false, false, true>
      <<<gC, blk256, 0, stream>>>(h_bf, Wvt, nullptr, nullptr, nullptr, v_bf);
  // 3) o = causal flash attention(q,k,v) -> bf16 [B,T,C]
  attn_kernel<<<dim3(kT / 64, kB * kNH), blk128, 0, stream>>>(q_bf, k_bf, v_bf, o_bf);
  // 4) x1 = x + o @ Wp + bp -> f32
  gemm_bf16_kernel<kC, kC, true, true, false, true, false>
      <<<gC, blk256, 0, stream>>>(o_bf, Wpt, bp, x, x1, nullptr);
  // 5) h2 = LN2(x1) -> bf16
  ln_kernel<<<kTok, blk256, 0, stream>>>(x1, ln2w, ln2b, h2_bf);
  // 6) a1 = relu(h2 @ W1 + b1) -> bf16
  gemm_bf16_kernel<kFF, kC, true, false, true, false, true>
      <<<gF, blk256, 0, stream>>>(h2_bf, W1t, b1, nullptr, nullptr, a1_bf);
  // 7) out = x1 + a1 @ W2 + b2 -> f32
  gemm_bf16_kernel<kC, kFF, true, true, false, true, false>
      <<<gC, blk256, 0, stream>>>(a1_bf, W2t, b2, x1, out, nullptr);
  (void)in_sizes; (void)n_in; (void)out_size; (void)ws_size;
}
